// OctreeAttention_27247272526289
// MI455X (gfx1250) — compile-verified, matrix operand order checked by reading, not run on hardware
//
#include <hip/hip_runtime.h>
#include <hip/hip_bf16.h>
#include <stdint.h>

// ---------------- problem constants ----------------
#define C_DIM      768
#define H_DIM      12
#define K_TOK      24      // tokens per attention sequence
#define D_DIL      4       // dilation
#define N_TOK      48100
#define BATCHFILL  8
#define MASK_VAL   -1000.0f
#define BLOCK_TOK  96              // K_TOK * D_DIL
#define NA         48192           // ceil(N/96)*96
#define NGROUPS    2008            // (NA/96)*4
#define QKV_COLS   2304            // 3*C

typedef __bf16 bf16;
typedef __attribute__((ext_vector_type(16))) __bf16 v16bf;
typedef __attribute__((ext_vector_type(8)))  __bf16 v8bf;
typedef __attribute__((ext_vector_type(8)))  float  v8f;

union Frag { v16bf v; unsigned u[8]; };

// K index pattern for 16-bit A/B fragments (CDNA5 ISA, 16x32 layout):
// dword d of lane: k = half*8 + (d<4 ? 2d : 16+2(d-4)), two consecutive bf16 packed.
__device__ __forceinline__ int frag_k(int half, int d) {
    return half * 8 + (d < 4 ? 2 * d : 16 + 2 * (d - 4));
}

// ---------------- fp32 -> bf16 convert (with zero row padding) ----------------
__global__ void cvt_pad_kernel(const float* __restrict__ src, bf16* __restrict__ dst,
                               long total, long validElems)
{
    long i = (long)blockIdx.x * blockDim.x + threadIdx.x;
    long stride = (long)gridDim.x * blockDim.x;
    for (; i < total; i += stride) {
        float v = (i < validElems) ? src[i] : 0.0f;
        dst[i] = (bf16)v;
    }
}

// ---------------- tiled WMMA GEMM: Out = A(bf16) @ B(bf16) + bias ----------------
// 128 threads = 4 waves; block tile 64x64; wave tile 32x32 (2x2 of 16x16); K-step 32.
template <bool F32OUT>
__global__ __launch_bounds__(128)
void wmma_gemm(const bf16* __restrict__ A, int lda, int Arows,
               const bf16* __restrict__ B, int ldb, int Kdim,
               const float* __restrict__ bias,
               void* __restrict__ Out, int ldc, int outRows)
{
    __shared__ bf16 lA[64][40];   // M x K tile (padded stride)
    __shared__ bf16 lBt[64][40];  // N x K tile (transposed so lane reads contiguous K-pairs)

    const int tid  = threadIdx.x;
    const int lane = tid & 31;
    const int wave = tid >> 5;
    const int half = lane >> 4;
    const int l15  = lane & 15;
    const int m0 = blockIdx.x * 64;
    const int n0 = blockIdx.y * 64;
    const int wm = (wave >> 1) * 32;
    const int wn = (wave & 1) * 32;

    const int ar = tid >> 1;        // A stage: row 0..63
    const int ac = (tid & 1) * 16;  // A stage: col 0 or 16
    const int bk = tid >> 2;        // B stage: k 0..31
    const int bs = (tid & 3) * 16;  // B stage: n segment

    v8f acc[2][2] = {};

    for (int kb = 0; kb < Kdim; kb += 32) {
        __syncthreads();
        // ---- stage A tile (row-major) ----
        {
            int grow = m0 + ar;
            if (grow < Arows) {
                const bf16* ap = A + (size_t)grow * lda + kb + ac;
                v8bf x0 = *(const v8bf*)(ap);
                v8bf x1 = *(const v8bf*)(ap + 8);
                *(v8bf*)&lA[ar][ac]     = x0;
                *(v8bf*)&lA[ar][ac + 8] = x1;
                if (kb + 32 < Kdim)  // prefetch next K-slice (global_prefetch_b8)
                    __builtin_prefetch(ap + 32, 0, 1);
            } else {
                v8bf z = {};
                *(v8bf*)&lA[ar][ac]     = z;
                *(v8bf*)&lA[ar][ac + 8] = z;
            }
        }
        // ---- stage B tile transposed: lBt[n][k] = B[kb+k][n0+n] ----
        {
            const bf16* brow = B + (size_t)(kb + bk) * ldb + n0 + bs;
            v8bf y0 = *(const v8bf*)(brow);
            v8bf y1 = *(const v8bf*)(brow + 8);
#pragma unroll
            for (int e = 0; e < 8; ++e) lBt[bs + e][bk] = y0[e];
#pragma unroll
            for (int e = 0; e < 8; ++e) lBt[bs + 8 + e][bk] = y1[e];
        }
        __syncthreads();

        // ---- fragment loads + 2x2 WMMA ----
        Frag af[2], bfr[2];
#pragma unroll
        for (int mt = 0; mt < 2; ++mt) {
            int r = wm + mt * 16 + l15;
#pragma unroll
            for (int d = 0; d < 8; ++d)
                af[mt].u[d] = *(const unsigned*)&lA[r][frag_k(half, d)];
        }
#pragma unroll
        for (int nt = 0; nt < 2; ++nt) {
            int c = wn + nt * 16 + l15;
#pragma unroll
            for (int d = 0; d < 8; ++d)
                bfr[nt].u[d] = *(const unsigned*)&lBt[c][frag_k(half, d)];
        }
#pragma unroll
        for (int mt = 0; mt < 2; ++mt)
#pragma unroll
            for (int nt = 0; nt < 2; ++nt)
                acc[mt][nt] = __builtin_amdgcn_wmma_f32_16x16x32_bf16(
                    false, af[mt].v, false, bfr[nt].v, (short)0, acc[mt][nt], false, false);
    }

    // ---- epilogue: bias add + store (C/D layout: m = 8*half + vgpr, n = lane&15) ----
#pragma unroll
    for (int mt = 0; mt < 2; ++mt)
#pragma unroll
        for (int nt = 0; nt < 2; ++nt) {
            int gcol = n0 + wn + nt * 16 + l15;
            float bv = bias[gcol];
#pragma unroll
            for (int i = 0; i < 8; ++i) {
                int gm = m0 + wm + mt * 16 + half * 8 + i;
                if (gm < outRows) {
                    float v = acc[mt][nt][i] + bv;
                    if (F32OUT) ((float*)Out)[(size_t)gm * ldc + gcol] = v;
                    else        ((bf16*)Out)[(size_t)gm * ldc + gcol] = (bf16)v;
                }
            }
        }
}

// ---------------- fused dilated-block attention ----------------
// One workgroup per (superblock, dilation); 4 waves; each wave handles 3 heads.
// Token t of this sequence lives at padded row  sb*96 + t*4 + d.
__global__ __launch_bounds__(128)
void attn_kernel(const bf16* __restrict__ qkv, const int* __restrict__ bidx,
                 bf16* __restrict__ aout)
{
    __shared__ int  bsh[32];
    __shared__ float Ssh[4][32][36];
    __shared__ bf16  Psh[4][32][40];
    __shared__ bf16  Vsh[4][64][36];   // V transposed: Vsh[n][k]

    const int g    = blockIdx.x;
    const int sb   = g >> 2;
    const int dd   = g & 3;
    const int tid  = threadIdx.x;
    const int w    = tid >> 5;
    const int lane = tid & 31;
    const int half = lane >> 4;
    const int l15  = lane & 15;
    const int rowbase = sb * BLOCK_TOK + dd;

    if (tid < 32) {
        int t = tid;
        int gr = rowbase + t * D_DIL;
        bsh[t] = (t < K_TOK && gr < N_TOK) ? bidx[gr] : BATCHFILL;
    }
    __syncthreads();

    for (int h = w; h < H_DIM; h += 4) {
        // ======== S = Q @ K^T  (24x24 padded to 32x32, hd=64 in two K-steps) ========
        v8f s[2][2] = {};
#pragma unroll
        for (int ks = 0; ks < 2; ++ks) {
            Frag aq[2], kf[2];
#pragma unroll
            for (int mt = 0; mt < 2; ++mt) {
                int t = mt * 16 + l15; if (t >= K_TOK) t = K_TOK - 1; // clamp pad rows
                const unsigned* p = (const unsigned*)
                    (qkv + (size_t)(rowbase + t * D_DIL) * QKV_COLS + h * 64 + ks * 32);
#pragma unroll
                for (int d = 0; d < 8; ++d) aq[mt].u[d] = p[frag_k(half, d) >> 1];
            }
#pragma unroll
            for (int nt = 0; nt < 2; ++nt) {
                int t = nt * 16 + l15; if (t >= K_TOK) t = K_TOK - 1;
                const unsigned* p = (const unsigned*)
                    (qkv + (size_t)(rowbase + t * D_DIL) * QKV_COLS + C_DIM + h * 64 + ks * 32);
#pragma unroll
                for (int d = 0; d < 8; ++d) kf[nt].u[d] = p[frag_k(half, d) >> 1];
            }
#pragma unroll
            for (int mt = 0; mt < 2; ++mt)
#pragma unroll
                for (int nt = 0; nt < 2; ++nt)
                    s[mt][nt] = __builtin_amdgcn_wmma_f32_16x16x32_bf16(
                        false, aq[mt].v, false, kf[nt].v, (short)0, s[mt][nt], false, false);
        }
        // dump S to LDS (D layout -> row/col)
#pragma unroll
        for (int mt = 0; mt < 2; ++mt)
#pragma unroll
            for (int nt = 0; nt < 2; ++nt)
#pragma unroll
                for (int i = 0; i < 8; ++i)
                    Ssh[w][mt * 16 + half * 8 + i][nt * 16 + l15] = s[mt][nt][i];
        __syncthreads();

        // ======== masked softmax, one row per lane; write P (bf16, zero-padded) ========
        {
            int row = lane;
            if (row < K_TOK) {
                int myb = bsh[row];
                float v[K_TOK];
                float mx = -3.0e38f;
#pragma unroll
                for (int j = 0; j < K_TOK; ++j) {
                    float x = Ssh[w][row][j] * 0.125f + ((bsh[j] != myb) ? MASK_VAL : 0.0f);
                    v[j] = x; mx = fmaxf(mx, x);
                }
                float sum = 0.0f;
#pragma unroll
                for (int j = 0; j < K_TOK; ++j) { float e = __expf(v[j] - mx); v[j] = e; sum += e; }
                float inv = 1.0f / sum;
#pragma unroll
                for (int j = 0; j < K_TOK; ++j) Psh[w][row][j] = (bf16)(v[j] * inv);
#pragma unroll
                for (int j = K_TOK; j < 32; ++j) Psh[w][row][j] = (bf16)0.0f;
            } else {
#pragma unroll
                for (int j = 0; j < 32; ++j) Psh[w][lane][j] = (bf16)0.0f;
            }
        }
        // ======== stage V transposed into LDS: Vsh[n][k] = V[k][n], zero pad k>=24 ========
        {
            int t = lane;
            if (t < K_TOK) {
                const bf16* vr = qkv + (size_t)(rowbase + t * D_DIL) * QKV_COLS + 2 * C_DIM + h * 64;
#pragma unroll
                for (int seg = 0; seg < 8; ++seg) {
                    v8bf vv = *(const v8bf*)(vr + seg * 8);
#pragma unroll
                    for (int e = 0; e < 8; ++e) Vsh[w][seg * 8 + e][t] = vv[e];
                }
            } else {
#pragma unroll
                for (int n = 0; n < 64; ++n) Vsh[w][n][t] = (bf16)0.0f;
            }
        }
        __syncthreads();

        // ======== O = P @ V  (32x64, single K-step of 32) ========
        Frag ap[2], bv[4];
#pragma unroll
        for (int mt = 0; mt < 2; ++mt) {
            int r = mt * 16 + l15;
#pragma unroll
            for (int d = 0; d < 8; ++d)
                ap[mt].u[d] = *(const unsigned*)&Psh[w][r][frag_k(half, d)];
        }
#pragma unroll
        for (int nt = 0; nt < 4; ++nt) {
            int c = nt * 16 + l15;
#pragma unroll
            for (int d = 0; d < 8; ++d)
                bv[nt].u[d] = *(const unsigned*)&Vsh[w][c][frag_k(half, d)];
        }
        v8f o[2][4] = {};
#pragma unroll
        for (int mt = 0; mt < 2; ++mt)
#pragma unroll
            for (int nt = 0; nt < 4; ++nt)
                o[mt][nt] = __builtin_amdgcn_wmma_f32_16x16x32_bf16(
                    false, ap[mt].v, false, bv[nt].v, (short)0, o[mt][nt], false, false);

        // store O back to padded layout (rows t<24 only)
#pragma unroll
        for (int mt = 0; mt < 2; ++mt)
#pragma unroll
            for (int nt = 0; nt < 4; ++nt)
#pragma unroll
                for (int i = 0; i < 8; ++i) {
                    int t = mt * 16 + half * 8 + i;
                    if (t < K_TOK)
                        aout[(size_t)(rowbase + t * D_DIL) * C_DIM + h * 64 + nt * 16 + l15] =
                            (bf16)o[mt][nt][i];
                }
        __syncthreads();
    }
}

// ---------------- host launcher ----------------
extern "C" void kernel_launch(void* const* d_in, const int* in_sizes, int n_in,
                              void* d_out, int out_size, void* d_ws, size_t ws_size,
                              hipStream_t stream)
{
    const float* data      = (const float*)d_in[0];
    const int*   batch_idx = (const int*)d_in[1];
    const float* w_qkv     = (const float*)d_in[2];
    const float* b_qkv     = (const float*)d_in[3];
    const float* w_proj    = (const float*)d_in[4];
    const float* b_proj    = (const float*)d_in[5];

    char* ws = (char*)d_ws;
    size_t off = 0;
    auto alloc = [&](size_t bytes) -> void* {
        off = (off + 255) & ~(size_t)255;
        void* p = ws + off;
        off += bytes;
        return p;
    };
    bf16* dataBf  = (bf16*)alloc((size_t)NA * C_DIM * sizeof(bf16));
    bf16* wqkvBf  = (bf16*)alloc((size_t)C_DIM * QKV_COLS * sizeof(bf16));
    bf16* wprojBf = (bf16*)alloc((size_t)C_DIM * C_DIM * sizeof(bf16));
    bf16* qkvBf   = (bf16*)alloc((size_t)NA * QKV_COLS * sizeof(bf16));
    bf16* attnBf  = (bf16*)alloc((size_t)NA * C_DIM * sizeof(bf16));
    (void)ws_size; (void)in_sizes; (void)n_in; (void)out_size;

    // fp32 -> bf16 (data zero-padded to NA rows)
    cvt_pad_kernel<<<2048, 256, 0, stream>>>(data,   dataBf,  (long)NA * C_DIM,      (long)N_TOK * C_DIM);
    cvt_pad_kernel<<<1024, 256, 0, stream>>>(w_qkv,  wqkvBf,  (long)C_DIM * QKV_COLS,(long)C_DIM * QKV_COLS);
    cvt_pad_kernel<<<512,  256, 0, stream>>>(w_proj, wprojBf, (long)C_DIM * C_DIM,   (long)C_DIM * C_DIM);

    // QKV GEMM: (NA x 768) @ (768 x 2304) + b_qkv  -> bf16
    dim3 g1(NA / 64, QKV_COLS / 64);
    wmma_gemm<false><<<g1, 128, 0, stream>>>(dataBf, C_DIM, NA,
                                             wqkvBf, QKV_COLS, C_DIM,
                                             b_qkv, qkvBf, QKV_COLS, NA);

    // fused dilated block attention -> bf16 (NA x 768)
    attn_kernel<<<NGROUPS, 128, 0, stream>>>(qkvBf, batch_idx, attnBf);

    // output projection: (N x 768) @ (768 x 768) + b_proj -> fp32 d_out
    dim3 g2((N_TOK + 63) / 64, C_DIM / 64);
    wmma_gemm<true><<<g2, 128, 0, stream>>>(attnBf, C_DIM, NA,
                                            wprojBf, C_DIM, C_DIM,
                                            b_proj, d_out, C_DIM, N_TOK);
}
